// constrained_network_46084999086876
// MI455X (gfx1250) — compile-verified, matrix operand order checked by reading, not run on hardware
//
#include <hip/hip_runtime.h>

// ---------------- problem constants (from reference) ----------------
#define NS_  32
#define NV_  16
#define NB_  8
#define D_   32
#define L_   2
#define HR_  64
#define NW_  3072          // NS*2*NV + NV*2*(NS+NV) + NV*2*NV
#define NSV_ 48            // NS + NV
#define TILE_E 16
#define NCT (NW_/16)       // 192 column tiles
#define PI_F 3.14159265358979f

typedef __bf16 bf16;
typedef __attribute__((ext_vector_type(16))) __bf16 bf16x16;
typedef __attribute__((ext_vector_type(8)))  float  f32x8;

union BFV { bf16x16 v; bf16 e[16]; };
union F8V { f32x8  v; float e[8]; };

__device__ __forceinline__ float silu_f(float x) {
  return x / (1.0f + __expf(-x));
}
__device__ __forceinline__ float sigm_f(float x) {
  return 1.0f / (1.0f + __expf(-x));
}

// ---------------- W_r2 [L][HR][NW] f32 -> w2t [L][NW][HR] bf16 ----------------
__global__ void k_prep(const float* __restrict__ Wr2, bf16* __restrict__ w2t) {
  int idx = blockIdx.x * 256 + threadIdx.x;
  if (idx < L_ * NW_ * HR_) {
    int l   = idx / (NW_ * HR_);
    int rem = idx % (NW_ * HR_);
    int col = rem / HR_;
    int k   = rem % HR_;
    w2t[idx] = (bf16)Wr2[(l * HR_ + k) * NW_ + col];
  }
}

// ---------------- QR (Gram-Schmidt) of W_uplift [NV,2] -> Q [NV,2] ----------------
__global__ void k_qr(const float* __restrict__ Wup, float* __restrict__ Q) {
  if (blockIdx.x == 0 && threadIdx.x == 0) {
    float c0[NV_], c1[NV_];
    for (int i = 0; i < NV_; ++i) { c0[i] = Wup[i*2+0]; c1[i] = Wup[i*2+1]; }
    float n0 = 0.f; for (int i = 0; i < NV_; ++i) n0 += c0[i]*c0[i];
    float inv0 = 1.f / sqrtf(fmaxf(n0, 1e-30f));
    for (int i = 0; i < NV_; ++i) c0[i] *= inv0;
    float d = 0.f; for (int i = 0; i < NV_; ++i) d += c0[i]*c1[i];
    for (int i = 0; i < NV_; ++i) c1[i] -= d * c0[i];
    float n1 = 0.f; for (int i = 0; i < NV_; ++i) n1 += c1[i]*c1[i];
    float inv1 = 1.f / sqrtf(fmaxf(n1, 1e-30f));
    for (int i = 0; i < NV_; ++i) { Q[i*2+0] = c0[i]; Q[i*2+1] = c1[i]*inv1; }
  }
}

// ---------------- uplift: y_v = Q x , y_s = 0 ; init both buffers ----------------
__global__ void __launch_bounds__(64) k_init(
    const float* __restrict__ x, const float* __restrict__ Q,
    float* __restrict__ ysA, float* __restrict__ yvA,
    float* __restrict__ ysB, float* __restrict__ yvB,
    float* __restrict__ x_cur) {
  int n = blockIdx.x, j = threadIdx.x;
  if (j < NV_*3) {
    int c = j / 3, k = j % 3;
    float v = Q[c*2+0]*x[n*6+k] + Q[c*2+1]*x[n*6+3+k];
    yvA[n*48+j] = v; yvB[n*48+j] = v;
  }
  if (j < NS_) { ysA[n*32+j] = 0.f; ysB[n*32+j] = 0.f; }
  if (j < 6)   { x_cur[n*6+j] = x[n*6+j]; }
}

// ---------------- per-node lin1 + zero scatter accumulators ----------------
__global__ void __launch_bounds__(64) k_pre(
    const float* __restrict__ ys, const float* __restrict__ yv,
    const float* __restrict__ Wl1s, const float* __restrict__ Wl1v,
    float* __restrict__ s1, float* __restrict__ v1,
    float* __restrict__ ns_acc, float* __restrict__ nv_acc) {
  int n = blockIdx.x, j = threadIdx.x;
  __shared__ float sy[NS_], sv[NV_*3];
  if (j < NS_)  sy[j] = ys[n*32+j];
  if (j < 48)   sv[j] = yv[n*48+j];
  __syncthreads();
  if (j < NS_) {
    float acc = 0.f;
    for (int c = 0; c < NS_; ++c) acc += sy[c] * Wl1s[c*NS_+j];
    s1[n*32+j] = acc * 0.17677670f;          // 1/sqrt(NS)
  }
  if (j < 48) {
    int co = j / 3, k = j % 3;
    float acc = 0.f;
    for (int c = 0; c < NV_; ++c) acc += sv[c*3+k] * Wl1v[c*NV_+co];
    v1[n*48+j] = acc * 0.25f;                // 1/sqrt(NV)
    ns_acc[n*48+j] = 0.f;
    nv_acc[n*48+j] = 0.f;
  }
}

// ---- B-fragment loader (16-bit B layout: lane = column, K halves split) ----
__device__ __forceinline__ void loadB(const bf16* __restrict__ w2t,
                                      int ct, int col, int koff,
                                      BFV& b0, BFV& b1) {
  const bf16* bp = &w2t[(ct*16 + col) * HR_];
  #pragma unroll
  for (int q = 0; q < 8; ++q) {
    b0.e[q]   = bp[koff + q];      b0.e[8+q] = bp[16 + koff + q];
    b1.e[q]   = bp[32 + koff + q]; b1.e[8+q] = bp[48 + koff + q];
  }
}

__device__ __forceinline__ f32x8 gemm64(const BFV& a0, const BFV& a1,
                                        const BFV& b0, const BFV& b1) {
  f32x8 d = (f32x8){0.f,0.f,0.f,0.f,0.f,0.f,0.f,0.f};
  d = __builtin_amdgcn_wmma_f32_16x16x32_bf16(false, a0.v, false, b0.v,
                                              (short)0, d, false, false);
  d = __builtin_amdgcn_wmma_f32_16x16x32_bf16(false, a1.v, false, b1.v,
                                              (short)0, d, false, false);
  return d;
}

// ---------------- fused edge kernel: radial MLP (WMMA) + TP + scatter ----------------
__global__ void __launch_bounds__(128) k_edge(
    const int*   __restrict__ esrc, const int* __restrict__ edst,
    const float* __restrict__ x_cur,        // [N][6]
    const float* __restrict__ s1,           // [N][NS]
    const float* __restrict__ v1,           // [N][NV][3]
    const float* __restrict__ Wr1,          // [16][64] layer slice
    const bf16*  __restrict__ w2t,          // [NW][64] bf16 layer slice
    float* __restrict__ ns_acc,             // [N][48]
    float* __restrict__ nv_acc)             // [N][48]
{
  __shared__ float sh_w1[16*64];
  __shared__ float sh_ea[TILE_E][2][3];
  __shared__ float sh_ef[TILE_E][16];
  __shared__ bf16  sh_hr[TILE_E][64];
  __shared__ float sh_ssrc[TILE_E][NS_];
  __shared__ float sh_dots[TILE_E][NV_][2];
  __shared__ float sh_crs[TILE_E][NV_][2][3];
  __shared__ float sh_es[TILE_E][48];
  __shared__ float sh_ev[TILE_E][NV_][3];
  __shared__ int   sh_src[TILE_E];
  __shared__ int   sh_dst[TILE_E];

  const int t  = threadIdx.x;
  const int e0 = blockIdx.x * TILE_E;

  for (int i = t; i < 16*64; i += 128) sh_w1[i] = Wr1[i];

  // stage 1: edge geometry, cutoff*SH attr, bessel feats   (32 threads: (e,m))
  if (t < TILE_E*2) {
    int e = t >> 1, m = t & 1;
    int gs = esrc[e0+e], gd = edst[e0+e];
    if (m == 0) { sh_src[e] = gs; sh_dst[e] = gd; }
    float vx = x_cur[gs*6 + m*3 + 0] - x_cur[gd*6 + m*3 + 0];
    float vy = x_cur[gs*6 + m*3 + 1] - x_cur[gd*6 + m*3 + 1];
    float vz = x_cur[gs*6 + m*3 + 2] - x_cur[gd*6 + m*3 + 2];
    float len = sqrtf(vx*vx + vy*vy + vz*vz + 1e-12f);
    float u = 2.0f * (len * (1.0f/3.0f) - 1.0f);
    float cut = 0.5f * (1.0f - __cosf(PI_F * u));
    if (u > 0.0f)  cut = 0.0f;
    if (u < -1.0f) cut = 1.0f;
    float s = 1.7320508f * cut / len;       // sqrt(3) * cutoff / len
    sh_ea[e][m][0] = s*vx; sh_ea[e][m][1] = s*vy; sh_ea[e][m][2] = s*vz;
    float lc = fmaxf(len, 1e-9f);
    float pref = 2.3094011f / lc;           // sqrt(2/3)*sqrt(8)/l
    #pragma unroll
    for (int b = 0; b < NB_; ++b)
      sh_ef[e][m*8+b] = pref * __sinf((float)(b+1) * PI_F * lc * (1.0f/3.0f));
  }
  __syncthreads();

  // stage 2a: hr = silu(ef @ W_r1 / sqrt(2B))  -> bf16 A operand in LDS
  {
    int e  = t >> 3;
    int ob = (t & 7) * 8;
    float ef[16];
    #pragma unroll
    for (int j = 0; j < 16; ++j) ef[j] = sh_ef[e][j];
    #pragma unroll
    for (int oo = 0; oo < 8; ++oo) {
      int o = ob + oo;
      float acc = 0.f;
      #pragma unroll
      for (int j = 0; j < 16; ++j) acc += ef[j] * sh_w1[j*64 + o];
      acc *= 0.25f;                          // 1/sqrt(16)
      sh_hr[e][o] = (bf16)silu_f(acc);
    }
  }
  // stage 2b: gather s1[src]
  for (int i = t; i < TILE_E*NS_; i += 128) {
    int e = i >> 5, c = i & 31;
    sh_ssrc[e][c] = s1[sh_src[e]*NS_ + c];
  }
  // stage 2c: dots + crosses of v1[src] with edge attrs
  for (int i = t; i < TILE_E*NV_; i += 128) {
    int e = i >> 4, c = i & 15;
    const float* vp = &v1[sh_src[e]*48 + c*3];
    float v0 = vp[0], v1_ = vp[1], v2 = vp[2];
    #pragma unroll
    for (int m = 0; m < 2; ++m) {
      float a0 = sh_ea[e][m][0], a1 = sh_ea[e][m][1], a2 = sh_ea[e][m][2];
      sh_dots[e][c][m]  = v0*a0 + v1_*a1 + v2*a2;
      sh_crs[e][c][m][0] = v1_*a2 - v2*a1;
      sh_crs[e][c][m][1] = v2*a0 - v0*a2;
      sh_crs[e][c][m][2] = v0*a1 - v1_*a0;
    }
  }
  // stage 2d: zero block accumulators
  for (int i = t; i < TILE_E*48; i += 128) {
    ((float*)sh_es)[i] = 0.f;
    ((float*)sh_ev)[i] = 0.f;
  }
  __syncthreads();

  // ---------------- main loop ----------------
  // wave index forced into an SGPR so all loop bounds / tile decode are scalar
  const int wave_u = __builtin_amdgcn_readfirstlane(t >> 5);
  const int lane = t & 31;
  const int hi   = lane >> 4;                // row-half per f32 D layout
  const int col  = lane & 15;                // output column of the tile
  const int koff = hi * 8;                   // K sub-offset per 16-bit A/B layout

  BFV a0u, a1u;                              // hr fragments (reused every tile)
  {
    const bf16* hp = &sh_hr[col][0];         // A: M = lane%16
    #pragma unroll
    for (int q = 0; q < 8; ++q) {
      a0u.e[q]   = hp[koff + q];      a0u.e[8+q] = hp[16 + koff + q];
      a1u.e[q]   = hp[32 + koff + q]; a1u.e[8+q] = hp[48 + koff + q];
    }
  }

  float acc_v[8][3], acc_s[8][3];
  #pragma unroll
  for (int r = 0; r < 8; ++r) {
    acc_v[r][0]=acc_v[r][1]=acc_v[r][2]=0.f;
    acc_s[r][0]=acc_s[r][1]=acc_s[r][2]=0.f;
  }

  // contiguous 48-tile range per wave, clipped against the three TP regions
  const int lo  = wave_u * 48, hiw = lo + 48;
  const int aHi = hiw < 64 ? hiw : 64;                         // sv region
  const int bLo = lo > 64 ? lo : 64;
  const int bHi = hiw < 160 ? hiw : 160;                       // vs region
  const int cLo = lo > 160 ? lo : 160;                         // vv region

  BFV b0c, b1c, b0n, b1n;

  // ---- region sv: w_sv (0e x 1o -> 1o) ----
  if (lo < aHi) {
    loadB(w2t, lo, col, koff, b0c, b1c);
    for (int ct = lo; ct < aHi; ++ct) {
      int nct = (ct + 1 < aHi) ? ct + 1 : ct;
      loadB(w2t, nct, col, koff, b0n, b1n);
      int pf = ct + 2 < NCT ? ct + 2 : NCT - 1;
      __builtin_prefetch(&w2t[(pf*16 + col) * HR_], 0, 1);
      F8V d; d.v = gemm64(a0u, a1u, b0c, b1c);
      int s = ct >> 1, m = ct & 1;           // SALU decode
      #pragma unroll
      for (int r = 0; r < 8; ++r) {
        int e = r + hi*8;
        float wgt = d.e[r] * sh_ssrc[e][s];
        acc_v[r][0] += wgt * sh_ea[e][m][0];
        acc_v[r][1] += wgt * sh_ea[e][m][1];
        acc_v[r][2] += wgt * sh_ea[e][m][2];
      }
      b0c = b0n; b1c = b1n;
    }
  }
  // ---- region vs: w_vs (1o . 1o -> 0e) ----
  if (bLo < bHi) {
    loadB(w2t, bLo, col, koff, b0c, b1c);
    for (int ct = bLo; ct < bHi; ++ct) {
      int nct = (ct + 1 < bHi) ? ct + 1 : ct;
      loadB(w2t, nct, col, koff, b0n, b1n);
      int pf = ct + 2 < NCT ? ct + 2 : NCT - 1;
      __builtin_prefetch(&w2t[(pf*16 + col) * HR_], 0, 1);
      F8V d; d.v = gemm64(a0u, a1u, b0c, b1c);
      int off = ct - 64;
      int c = off / 6, rem = off % 6;
      int m = rem / 3, slot = rem % 3;       // SALU decode
      #pragma unroll
      for (int r = 0; r < 8; ++r) {
        int e = r + hi*8;
        acc_s[r][slot] += d.e[r] * sh_dots[e][c][m];
      }
      b0c = b0n; b1c = b1n;
    }
  }
  // ---- region vv: w_vv (1o x 1o -> 1o, cross) ----
  if (cLo < hiw) {
    loadB(w2t, cLo, col, koff, b0c, b1c);
    for (int ct = cLo; ct < hiw; ++ct) {
      int nct = (ct + 1 < hiw) ? ct + 1 : ct;
      loadB(w2t, nct, col, koff, b0n, b1n);
      int pf = ct + 2 < NCT ? ct + 2 : NCT - 1;
      __builtin_prefetch(&w2t[(pf*16 + col) * HR_], 0, 1);
      F8V d; d.v = gemm64(a0u, a1u, b0c, b1c);
      int off = ct - 160;
      int c = off >> 1, m = off & 1;         // SALU decode
      #pragma unroll
      for (int r = 0; r < 8; ++r) {
        int e = r + hi*8;
        acc_v[r][0] += d.e[r] * sh_crs[e][c][m][0];
        acc_v[r][1] += d.e[r] * sh_crs[e][c][m][1];
        acc_v[r][2] += d.e[r] * sh_crs[e][c][m][2];
      }
      b0c = b0n; b1c = b1n;
    }
  }

  // fold scales: w*(1/sqrt HR), TP norms, segment 1/sqrt(NUM_NEIGHBORS)
  const float FV = 0.125f * 0.125f    * 0.25f;   // sv & vv share 1/sqrt(64)
  const float FS = 0.125f * 0.10206207f * 0.25f; // 1/sqrt(96)
  #pragma unroll
  for (int r = 0; r < 8; ++r) {
    int e = r + hi*8;
    atomicAdd(&sh_ev[e][col][0], acc_v[r][0] * FV);
    atomicAdd(&sh_ev[e][col][1], acc_v[r][1] * FV);
    atomicAdd(&sh_ev[e][col][2], acc_v[r][2] * FV);
    atomicAdd(&sh_es[e][ 0 + col], acc_s[r][0] * FS);
    atomicAdd(&sh_es[e][16 + col], acc_s[r][1] * FS);
    atomicAdd(&sh_es[e][32 + col], acc_s[r][2] * FS);
  }
  __syncthreads();

  // scatter-aggregate to destination nodes
  for (int i = t; i < TILE_E*48; i += 128) {
    int e = i / 48, j = i % 48;
    int n = sh_dst[e];
    atomicAdd(&ns_acc[n*48 + j], sh_es[e][j]);
    atomicAdd(&nv_acc[n*48 + j], ((float*)sh_ev)[e*48 + j]);
  }
}

// ---------------- per-node: self-connection, lin2, gate, SI, leapfrog, project ----------------
__global__ void __launch_bounds__(64) k_post(
    const int*   __restrict__ node_attr, const float* __restrict__ emb,
    const float* __restrict__ ys,  const float* __restrict__ yv,
    const float* __restrict__ ys_old, const float* __restrict__ yv_old,
    const float* __restrict__ ns_acc, const float* __restrict__ nv_acc,
    const float* __restrict__ Wscs, const float* __restrict__ Wscv,
    const float* __restrict__ Wl2s, const float* __restrict__ Wl2v,
    const float* __restrict__ Wsis, const float* __restrict__ Wsiv,
    const float* __restrict__ harr, const float* __restrict__ mixarr, int layer,
    const float* __restrict__ Q,
    float* __restrict__ ys_new, float* __restrict__ yv_new,
    float* __restrict__ x_cur)
{
  int n = blockIdx.x, j = threadIdx.x;
  __shared__ float sa[D_], sy[NS_], svv[48], sns[48], snv[48], sconv[48], snewv[48];
  float hv = harr[layer];
  float h2 = hv * hv;
  float mv = mixarr[layer];
  if (j < D_)  sa[j] = emb[node_attr[n]*D_ + j];
  if (j < NS_) sy[j] = ys[n*32+j];
  if (j < 48)  { svv[j] = yv[n*48+j]; sns[j] = ns_acc[n*48+j]; snv[j] = nv_acc[n*48+j]; }
  __syncthreads();
  if (j < 48) {
    float acc = 0.f;                                  // sc_s[j]
    for (int i2 = 0; i2 < NS_; ++i2) {
      float yi = sy[i2];
      const float* wp = &Wscs[(i2*D_)*NSV_ + j];
      for (int a2 = 0; a2 < D_; ++a2) acc += yi * sa[a2] * wp[a2*NSV_];
    }
    float s2 = 0.f;
    for (int c = 0; c < 48; ++c) s2 += sns[c] * Wl2s[c*48 + j];
    sconv[j] = acc * 0.03125f + s2 * 0.14433757f;     // 1/sqrt(NS*D), 1/sqrt(48)
  }
  __syncthreads();
  if (j < NS_) {
    float gs = silu_f(sconv[j]);
    float si = 0.f;
    for (int c = 0; c < NS_; ++c) si += sy[c] * Wsis[c*NS_ + j];
    si *= 0.17677670f;
    ys_new[n*32+j] = 2.f*sy[j] - ys_old[n*32+j] + h2*(mv*gs + (mv-1.f)*si);
  }
  if (j < 48) {
    int co = j / 3, k = j % 3;
    float acc = 0.f;                                  // sc_v[co][k]
    for (int cc = 0; cc < NV_; ++cc) {
      float yvk = svv[cc*3+k];
      const float* wp = &Wscv[(cc*D_)*NV_ + co];
      for (int a2 = 0; a2 < D_; ++a2) acc += yvk * sa[a2] * wp[a2*NV_];
    }
    float v2 = 0.f;
    for (int cc = 0; cc < NV_; ++cc) v2 += snv[cc*3+k] * Wl2v[cc*NV_ + co];
    float conv = acc * 0.044194174f + v2 * 0.25f;     // 1/sqrt(NV*D), 1/sqrt(NV)
    float gv = sigm_f(sconv[32 + co]) * conv;
    float si = 0.f;
    for (int cc = 0; cc < NV_; ++cc) si += svv[cc*3+k] * Wsiv[cc*NV_ + co];
    si *= 0.25f;
    float nv2 = 2.f*svv[j] - yv_old[n*48+j] + h2*(mv*gv + (mv-1.f)*si);
    yv_new[n*48+j] = nv2;
    snewv[j] = nv2;
  }
  __syncthreads();
  if (j < 6) {                                        // x = Q^T y_v
    int i2 = j / 3, k = j % 3;
    float acc = 0.f;
    for (int c = 0; c < NV_; ++c) acc += Q[c*2 + i2] * snewv[c*3 + k];
    x_cur[n*6 + i2*3 + k] = acc;
  }
}

// ---------------------------------------------------------------------------
extern "C" void kernel_launch(void* const* d_in, const int* in_sizes, int n_in,
                              void* d_out, int out_size, void* d_ws, size_t ws_size,
                              hipStream_t stream) {
  const float* x         = (const float*)d_in[0];
  const int*   node_attr = (const int*)  d_in[2];
  const int*   esrc      = (const int*)  d_in[3];
  const int*   edst      = (const int*)  d_in[4];
  const float* emb       = (const float*)d_in[5];
  const float* Wup       = (const float*)d_in[6];
  const float* Wscs      = (const float*)d_in[7];
  const float* Wscv      = (const float*)d_in[8];
  const float* Wl1s      = (const float*)d_in[9];
  const float* Wl1v      = (const float*)d_in[10];
  const float* Wr1       = (const float*)d_in[11];
  const float* Wr2       = (const float*)d_in[12];
  const float* Wl2s      = (const float*)d_in[13];
  const float* Wl2v      = (const float*)d_in[14];
  const float* Wsis      = (const float*)d_in[15];
  const float* Wsiv      = (const float*)d_in[16];
  const float* harr      = (const float*)d_in[17];
  const float* mixarr    = (const float*)d_in[18];

  const int N = in_sizes[1];        // batch has N elements
  const int E = in_sizes[3];        // edge_src has E elements

  // workspace carve-up (256B aligned)
  char* ws = (char*)d_ws;
  auto carve = [&](size_t bytes) -> char* {
    char* p = ws; ws += (bytes + 255) & ~(size_t)255; return p;
  };
  float* Q     = (float*)carve(NV_*2*sizeof(float));
  float* x_cur = (float*)carve((size_t)N*6*sizeof(float));
  float* ysA   = (float*)carve((size_t)N*32*sizeof(float));
  float* yvA   = (float*)carve((size_t)N*48*sizeof(float));
  float* ysB   = (float*)carve((size_t)N*32*sizeof(float));
  float* yvB   = (float*)carve((size_t)N*48*sizeof(float));
  float* s1    = (float*)carve((size_t)N*32*sizeof(float));
  float* v1    = (float*)carve((size_t)N*48*sizeof(float));
  float* nsac  = (float*)carve((size_t)N*48*sizeof(float));
  float* nvac  = (float*)carve((size_t)N*48*sizeof(float));
  bf16*  w2t   = (bf16*) carve((size_t)L_*NW_*HR_*sizeof(bf16));

  k_prep<<<(L_*NW_*HR_ + 255)/256, 256, 0, stream>>>(Wr2, w2t);
  k_qr  <<<1, 32, 0, stream>>>(Wup, Q);
  k_init<<<N, 64, 0, stream>>>(x, Q, ysA, yvA, ysB, yvB, x_cur);

  float *cur_s = ysA, *cur_v = yvA, *old_s = ysB, *old_v = yvB;
  for (int l = 0; l < L_; ++l) {
    k_pre<<<N, 64, 0, stream>>>(cur_s, cur_v, Wl1s + l*NS_*NS_, Wl1v + l*NV_*NV_,
                                s1, v1, nsac, nvac);
    k_edge<<<E / TILE_E, 128, 0, stream>>>(esrc, edst, x_cur, s1, v1,
                                           Wr1 + l*16*HR_, w2t + (size_t)l*NW_*HR_,
                                           nsac, nvac);
    k_post<<<N, 64, 0, stream>>>(node_attr, emb, cur_s, cur_v, old_s, old_v,
                                 nsac, nvac,
                                 Wscs + l*NS_*D_*NSV_, Wscv + l*NV_*D_*NV_,
                                 Wl2s + l*NSV_*NSV_,   Wl2v + l*NV_*NV_,
                                 Wsis + l*NS_*NS_,     Wsiv + l*NV_*NV_,
                                 harr, mixarr, l, Q,
                                 old_s, old_v,   // new y overwrites old buffers
                                 x_cur);
    float* ts = cur_s; cur_s = old_s; old_s = ts;
    float* tv = cur_v; cur_v = old_v; old_v = tv;
  }
  hipMemcpyAsync(d_out, x_cur, (size_t)N*6*sizeof(float),
                 hipMemcpyDeviceToDevice, stream);
}
